// AnotherDDoIGRUCell_43181601194450
// MI455X (gfx1250) — matrix operand order repaired and checked by reading, not compiled, not run on hardware
//
#include <hip/hip_runtime.h>

// ---------------------------------------------------------------------------
// DDoIGRU cell for MI455X (gfx1250, wave32, WMMA).
// Phase 0: fold 7 input-weight blocks into time-shift GEMM weight Wcat(384x768)
//          and pack Wcat + hidden weights (256x768) into bf16 B-fragment layout.
// Phase 1: PRE = concat(x_t, x_{t-1}, x_{t-2}) @ Wcat  (131072 x 768, f32),
//          software-pipelined (double-buffered A/B fragments around WMMA).
// Phase 2: 8 persistent workgroups (16 batches each), h in LDS as bf16,
//          per-step (16x256)@(256x768) via v_wmma_f32_16x16x32_bf16 with
//          pipelined fragment loads, fused sigmoid/tanh gates, 1024 steps.
// ---------------------------------------------------------------------------

typedef __attribute__((ext_vector_type(16))) __bf16 bf16x16;
typedef __attribute__((ext_vector_type(8)))  __bf16 bf16x8;
typedef __attribute__((ext_vector_type(8)))  float  f32x8;

#define SEQ   1024
#define BATCH 128
#define IND   128
#define U     256
#define N3    768     // 3*U
#define KCAT  384     // 3*IND

// workspace layout (bytes)
#define WPACK_BYTES   (12*48*32*16*2)              // 589824
#define WHPACK_BYTES  (8*48*32*16*2)               // 393216
#define PRE_OFF       (WPACK_BYTES + WHPACK_BYTES) // 983040 (256-aligned)

#define KT_STRIDE     (48*32*16)                   // elems between kt planes

__device__ __forceinline__ float sigm(float x) { return 1.0f / (1.0f + __expf(-x)); }

// ---------------------------------------------------------------------------
// Pack folded input weights into B-fragment layout:
//   frag(kt,nt): 32 lanes x 16 bf16, lane -> N = nt*16 + (lane&15),
//   element e -> K = kt*32 + (e&7) + (e>=8 ? 16 : 0) + (lane>=16 ? 8 : 0)
// ---------------------------------------------------------------------------
__global__ __launch_bounds__(256) void pack_input_kernel(
    const float* __restrict__ iw, __bf16* __restrict__ wpack) {
  const int idx  = blockIdx.x * 256 + threadIdx.x;   // < 294912
  const int e    = idx & 15;
  const int lane = (idx >> 4) & 31;
  const int tile = idx >> 9;
  const int nt   = tile % 48;
  const int kt   = tile / 48;
  const int n    = nt * 16 + (lane & 15);
  const int k    = kt * 32 + (e & 7) + ((e & 8) ? 16 : 0) + ((lane & 16) ? 8 : 0);
  const int kb   = k >> 7;      // 0: x_t, 1: x_{t-1}, 2: x_{t-2}
  const int kr   = k & 127;
  const int s    = n >> 8;      // 0: r, 1: u, 2: h
  const int j    = n & 255;
  const float* row = iw + (size_t)kr * 1792;
  float v;
  if (s < 2) {
    const int off = s * 256;
    const float w  = row[off + j];           // Wxr / Wxu
    const float dw = row[768 + off + j];     // dWxr / dWxu
    const float dd = row[1280 + off + j];    // ddWxr / ddWxu
    v = (kb == 0) ? (w + dw + dd)
      : (kb == 1) ? -(dw + 2.0f * dd)
                  : dd;
  } else {
    v = (kb == 0) ? row[512 + j] : 0.0f;     // Wxh only on x_t block
  }
  wpack[idx] = (__bf16)v;
}

__global__ __launch_bounds__(256) void pack_hidden_kernel(
    const float* __restrict__ hw, __bf16* __restrict__ whp) {
  const int idx  = blockIdx.x * 256 + threadIdx.x;   // < 98304
  const int e    = idx & 15;
  const int lane = (idx >> 4) & 31;
  const int tile = idx >> 9;
  const int nt   = tile % 48;
  const int kt   = tile / 48;
  const int n    = nt * 16 + (lane & 15);
  const int k    = kt * 32 + (e & 7) + ((e & 8) ? 16 : 0) + ((lane & 16) ? 8 : 0);
  whp[idx] = (__bf16)hw[(size_t)k * N3 + n];
}

// ---------------------------------------------------------------------------
// Phase 1: PRE (131072 x 768) = Xcat (131072 x 384) @ Wcat (384 x 768)
// ---------------------------------------------------------------------------

// One A fragment (16 bf16/lane) = two 8-float spans: K0..K0+7, K0+16..K0+23.
// Both spans always lie in the same 128-wide time-shift block.
template <bool GUARD>
__device__ __forceinline__ void load_a16(float* f, const float* __restrict__ x,
                                         int b, int t, int K0) {
  const int blk = K0 >> 7;          // time shift (wave-uniform)
  const int col = K0 & 127;
  const int tt  = t - blk;
  if (!GUARD || tt >= 0) {
    const float* p = x + (((size_t)b << 10) + (size_t)tt) * IND + col;
    const float4 q0 = *reinterpret_cast<const float4*>(p);
    const float4 q1 = *reinterpret_cast<const float4*>(p + 4);
    const float4 q2 = *reinterpret_cast<const float4*>(p + 16);
    const float4 q3 = *reinterpret_cast<const float4*>(p + 20);
    f[0]=q0.x;  f[1]=q0.y;  f[2]=q0.z;  f[3]=q0.w;
    f[4]=q1.x;  f[5]=q1.y;  f[6]=q1.z;  f[7]=q1.w;
    f[8]=q2.x;  f[9]=q2.y;  f[10]=q2.z; f[11]=q2.w;
    f[12]=q3.x; f[13]=q3.y; f[14]=q3.z; f[15]=q3.w;
  } else {
#pragma unroll
    for (int i = 0; i < 16; ++i) f[i] = 0.0f;
  }
}

__device__ __forceinline__ void load_b4(bf16x16* dst, const __bf16* __restrict__ wb0,
                                        int kt) {
  const __bf16* p = wb0 + (size_t)kt * KT_STRIDE;
#pragma unroll
  for (int i = 0; i < 4; ++i)
    dst[i] = *reinterpret_cast<const bf16x16*>(p + (size_t)i * 512);
}

template <bool GUARD>
__device__ __forceinline__ void gemm_body(const float* __restrict__ x,
                                          const __bf16* __restrict__ wb0,
                                          int b, int t, int kb, f32x8* acc) {
  float   f[2][16];
  bf16x16 bb[2][4];
  load_a16<GUARD>(f[0], x, b, t, kb);      // kt = 0
  load_b4(bb[0], wb0, 0);
#pragma unroll
  for (int kt = 0; kt < 12; ++kt) {
    const int cur = kt & 1;
    const int nxt = cur ^ 1;
    if (kt < 11) {                         // prefetch kt+1 while kt computes
      load_a16<GUARD>(f[nxt], x, b, t, (kt + 1) * 32 + kb);
      load_b4(bb[nxt], wb0, kt + 1);
    }
    bf16x16 a;
#pragma unroll
    for (int i = 0; i < 16; ++i) a[i] = (__bf16)f[cur][i];
#pragma unroll
    for (int i = 0; i < 4; ++i)
      acc[i] = __builtin_amdgcn_wmma_f32_16x16x32_bf16(
          false, a, false, bb[cur][i], (short)0, acc[i], false, false);
  }
}

__global__ __launch_bounds__(256) void gemm_pre_kernel(
    const float* __restrict__ x, const __bf16* __restrict__ wpack,
    float* __restrict__ pre) {
  const int lane = threadIdx.x & 31;
  const int wave = threadIdx.x >> 5;   // 0..7
  const int mt   = wave & 3;           // M tile within WG
  const int nq   = wave >> 2;          // N quad (4 tiles each)
  const int rowTileBase = blockIdx.y * 64 + mt * 16;
  const int m = rowTileBase + (lane & 15);
  const int b = m >> 10;
  const int t = m & 1023;
  const int kb = (lane & 16) ? 8 : 0;  // A-frag K sub-offset
  const int ntBase = blockIdx.x * 8 + nq * 4;
  const __bf16* wb0 = wpack + ((size_t)ntBase * 32 + lane) * 16;

  f32x8 acc[4] = {};
  if ((rowTileBase & 1023) == 0)       // only tiles touching t<2 need guards
    gemm_body<true >(x, wb0, b, t, kb, acc);
  else
    gemm_body<false>(x, wb0, b, t, kb, acc);

  const int rowS = rowTileBase + ((lane & 16) ? 8 : 0);
#pragma unroll
  for (int i = 0; i < 4; ++i) {
    const int n = (ntBase + i) * 16 + (lane & 15);
#pragma unroll
    for (int v = 0; v < 8; ++v)
      pre[(size_t)(rowS + v) * N3 + n] = acc[i][v];
  }
}

// ---------------------------------------------------------------------------
// Phase 2: recurrent. 8 WGs x 768 threads (24 waves). Each WG owns 16 batches.
// ---------------------------------------------------------------------------
#define HSTRIDE 264   // bf16 row stride for h in LDS (bank-spread)
#define ASTRIDE 772   // f32  row stride for acc in LDS (bank-spread)

__global__ __launch_bounds__(768) void recurrent_kernel(
    const __bf16* __restrict__ whp, const float* __restrict__ pre,
    const float* __restrict__ bias, float* __restrict__ out) {
  __shared__ __bf16 hbuf[16][HSTRIDE];
  __shared__ float  accbuf[16][ASTRIDE];
  __shared__ float  biasS[N3];

  const int tid    = threadIdx.x;
  const int lane   = tid & 31;
  const int wave   = tid >> 5;          // 0..23
  const int batch0 = blockIdx.x * 16;

  for (int i = tid; i < 16 * HSTRIDE; i += 768)
    (&hbuf[0][0])[i] = (__bf16)0.0f;    // h0 = 0
  biasS[tid] = bias[tid];
  __syncthreads();

  const int mrow = lane & 15;
  const int kb   = (lane & 16) ? 8 : 0;
  const int mhi  = (lane & 16) ? 8 : 0;
  const int nt0  = wave * 2;            // 2 N-tiles per wave -> 48 tiles
  const __bf16* wbW = whp + ((size_t)nt0 * 32 + lane) * 16;  // + kt*24576

  for (int t = 0; t < SEQ; ++t) {
    f32x8 acc0 = {};
    f32x8 acc1 = {};

    bf16x8  lo[2], hi[2];
    bf16x16 w0[2], w1[2];
    // prologue: kt = 0 fragments
    lo[0] = *reinterpret_cast<const bf16x8*>(&hbuf[mrow][kb]);
    hi[0] = *reinterpret_cast<const bf16x8*>(&hbuf[mrow][kb + 16]);
    w0[0] = *reinterpret_cast<const bf16x16*>(wbW);
    w1[0] = *reinterpret_cast<const bf16x16*>(wbW + 512);

#pragma unroll
    for (int kt = 0; kt < 8; ++kt) {
      const int cur = kt & 1;
      const int nxt = cur ^ 1;
      if (kt < 7) {                     // prefetch kt+1 while kt computes
        const int k1 = (kt + 1) * 32 + kb;
        lo[nxt] = *reinterpret_cast<const bf16x8*>(&hbuf[mrow][k1]);
        hi[nxt] = *reinterpret_cast<const bf16x8*>(&hbuf[mrow][k1 + 16]);
        const __bf16* wn = wbW + (size_t)(kt + 1) * KT_STRIDE;
        w0[nxt] = *reinterpret_cast<const bf16x16*>(wn);
        w1[nxt] = *reinterpret_cast<const bf16x16*>(wn + 512);
      }
      bf16x16 a;
#pragma unroll
      for (int i = 0; i < 8; ++i) { a[i] = lo[cur][i]; a[8 + i] = hi[cur][i]; }
      acc0 = __builtin_amdgcn_wmma_f32_16x16x32_bf16(
          false, a, false, w0[cur], (short)0, acc0, false, false);
      acc1 = __builtin_amdgcn_wmma_f32_16x16x32_bf16(
          false, a, false, w1[cur], (short)0, acc1, false, false);
    }

    {
      const int n0 = nt0 * 16 + (lane & 15);
#pragma unroll
      for (int v = 0; v < 8; ++v) {
        accbuf[mhi + v][n0]      = acc0[v];
        accbuf[mhi + v][n0 + 16] = acc1[v];
      }
    }
    __syncthreads();

    for (int i = tid; i < 16 * U; i += 768) {
      const int bl = i >> 8;
      const int j  = i & 255;
      const size_t prow = ((size_t)(batch0 + bl) << 10) + (size_t)t;
      const float* pr = pre + prow * N3;
      const float r = sigm(accbuf[bl][j]       + pr[j]       + biasS[j]);
      const float u = sigm(accbuf[bl][U + j]   + pr[U + j]   + biasS[U + j]);
      const float c = tanhf(r * accbuf[bl][2*U + j] + pr[2*U + j] + biasS[2*U + j]);
      const float hold = (float)hbuf[bl][j];
      const float hn = u * hold + (1.0f - u) * c;
      out[prow * U + j] = hn;
      hbuf[bl][j] = (__bf16)hn;
    }
    __syncthreads();
  }
}

// ---------------------------------------------------------------------------
extern "C" void kernel_launch(void* const* d_in, const int* in_sizes, int n_in,
                              void* d_out, int out_size, void* d_ws, size_t ws_size,
                              hipStream_t stream) {
  const float* x    = (const float*)d_in[0];   // (128,1024,128)
  const float* iw   = (const float*)d_in[1];   // (128,1792)
  const float* hw   = (const float*)d_in[2];   // (256,768)
  const float* bias = (const float*)d_in[3];   // (768,)
  // d_in[4] (constant) is [I;0] by construction: shifts are hardcoded.
  float* out = (float*)d_out;

  char* ws = (char*)d_ws;
  __bf16* wpack = (__bf16*)ws;
  __bf16* whp   = (__bf16*)(ws + WPACK_BYTES);
  float*  pre   = (float*)(ws + PRE_OFF);

  pack_input_kernel <<<1152, 256, 0, stream>>>(iw, wpack);
  pack_hidden_kernel<<<384,  256, 0, stream>>>(hw, whp);

  dim3 g1(6, 2048);  // N blocks of 128, M blocks of 64 over 131072 rows
  gemm_pre_kernel<<<g1, 256, 0, stream>>>(x, wpack, pre);

  recurrent_kernel<<<8, 768, 0, stream>>>(whp, pre, bias, out);
}
